// TopKGate_64888365908349
// MI455X (gfx1250) — matrix operand order repaired and checked by reading, not compiled, not run on hardware
//
#include <hip/hip_runtime.h>
#include <hip/hip_bf16.h>
#include <math.h>

// ---------------------------------------------------------------------------
// MoE Top-2 gate for MI455X (gfx1250):
//   logits GEMM via v_wmma_f32_16x16x32_bf16, gate_w staged with
//   global_load_async_to_lds_b128 (ASYNCcnt path) overlapped with WMMA,
//   fused top-2 + renormalized softmax + scatter + aux-loss reduction.
// ---------------------------------------------------------------------------

typedef __attribute__((ext_vector_type(16))) __bf16 v16bf;
typedef __attribute__((ext_vector_type(8)))  float  v8f;
typedef __attribute__((ext_vector_type(4)))  int    v4i;

#define TOKENS          16384
#define DMODEL          2048
#define EXPERTS         64
#define ROWS_PER_BLOCK  128
#define KC              64      // K chunk staged in LDS per iteration
#define KCP             80      // padded row stride (bf16 elems) for Bt
#define LSTR            68      // padded logits row stride (floats)
#define NCHUNK          (DMODEL / KC)

// Issue async DMA of gate_w[kc0 : kc0+KC, 0:64] (contiguous 16 KB) into LDS.
__device__ __forceinline__
void issue_gw_async(const float* __restrict__ gw, float* raw, int kc0, int tid)
{
    const float* gsrc = gw + kc0 * EXPERTS;     // contiguous KC*64 floats
    #pragma unroll
    for (int p = 0; p < 4; ++p) {
        const int off = p * 1024 + tid * 4;     // floats; 16B per thread per op
        __builtin_amdgcn_global_load_async_to_lds_b128(
            (v4i*)(gsrc + off), (v4i*)(raw + off), 0, 0);
    }
}

// LDS->LDS transpose + fp32->bf16: raw[kl][n] -> Bt[n][kl]
__device__ __forceinline__
void transpose_gw(const float* raw, __bf16* Bt, int tid)
{
    #pragma unroll
    for (int i = tid; i < EXPERTS * (KC / 2); i += 256) {
        const int n  = i & 63;
        const int kp = i >> 6;                  // pair of K rows
        const float v0 = raw[(2 * kp)     * EXPERTS + n];
        const float v1 = raw[(2 * kp + 1) * EXPERTS + n];
        __bf16* bp = &Bt[n * KCP + 2 * kp];
        bp[0] = (__bf16)v0;                     // adjacent b16 stores -> merged
        bp[1] = (__bf16)v1;
    }
}

__global__ __launch_bounds__(256)
void moe_gate_kernel(const float* __restrict__ x,
                     const float* __restrict__ gw,
                     const float* __restrict__ gb,
                     float* __restrict__ gate_weights,
                     int*   __restrict__ sel_experts,
                     float* __restrict__ ws_prob,
                     float* __restrict__ ws_cnt)
{
    __shared__ __align__(16) float  raw[KC * EXPERTS];       // 16.0 KB (async landing)
    __shared__ __align__(32) __bf16 Bt[EXPERTS * KCP];       // 10.0 KB (transposed bf16)
    __shared__ float logitsLds[ROWS_PER_BLOCK * LSTR];       // 34.0 KB
    __shared__ float probSum[EXPERTS];
    __shared__ float cntSum[EXPERTS];

    const int tid  = threadIdx.x;
    const int wave = tid >> 5;
    const int lane = tid & 31;
    const int half = lane >> 4;     // 0: lanes 0-15, 1: lanes 16-31
    const int l16  = lane & 15;
    const int rowBase = blockIdx.x * ROWS_PER_BLOCK;
    const int myRow   = rowBase + wave * 16 + l16;   // A-matrix row this lane feeds

    if (tid < EXPERTS) { probSum[tid] = 0.0f; cntSum[tid] = 0.0f; }

    v8f acc0 = {}, acc1 = {}, acc2 = {}, acc3 = {};  // 4 N-tiles of 16 experts

    // ---- prologue: stage chunk 0
    issue_gw_async(gw, raw, 0, tid);
    asm volatile("s_wait_asynccnt 0x0" ::: "memory");
    __syncthreads();
    transpose_gw(raw, Bt, tid);
    __syncthreads();

    for (int chunk = 0; chunk < NCHUNK; ++chunk) {
        const int kc0 = chunk * KC;

        // Overlap next chunk's DMA with this chunk's WMMA work
        // (raw buffer is free: its contents were transposed into Bt already).
        if (chunk + 1 < NCHUNK)
            issue_gw_async(gw, raw, kc0 + KC, tid);

        #pragma unroll
        for (int kk = 0; kk < KC / 32; ++kk) {
            const int kglob = kc0 + kk * 32;
            // ---- A fragment: 16x32 bf16; lane holds row myRow,
            //      K = half*16 .. +15 (contiguous, matches B)
            const float* xp = &x[myRow * DMODEL + kglob + half * 16];
            const float4 f0 = ((const float4*)xp)[0];
            const float4 f1 = ((const float4*)xp)[1];
            const float4 f2 = ((const float4*)(xp + 8))[0];
            const float4 f3 = ((const float4*)(xp + 8))[1];
            v16bf a;
            a[0]=(__bf16)f0.x;  a[1]=(__bf16)f0.y;  a[2]=(__bf16)f0.z;  a[3]=(__bf16)f0.w;
            a[4]=(__bf16)f1.x;  a[5]=(__bf16)f1.y;  a[6]=(__bf16)f1.z;  a[7]=(__bf16)f1.w;
            a[8]=(__bf16)f2.x;  a[9]=(__bf16)f2.y;  a[10]=(__bf16)f2.z; a[11]=(__bf16)f2.w;
            a[12]=(__bf16)f3.x; a[13]=(__bf16)f3.y; a[14]=(__bf16)f3.z; a[15]=(__bf16)f3.w;

            // ---- B fragments: 32x16 bf16 per N-tile; column = l16, K-half = half*16
            const int kfrag = kk * 32 + half * 16;
            const v16bf b0 = *(const v16bf*)&Bt[( 0 + l16) * KCP + kfrag];
            const v16bf b1 = *(const v16bf*)&Bt[(16 + l16) * KCP + kfrag];
            const v16bf b2 = *(const v16bf*)&Bt[(32 + l16) * KCP + kfrag];
            const v16bf b3 = *(const v16bf*)&Bt[(48 + l16) * KCP + kfrag];

            acc0 = __builtin_amdgcn_wmma_f32_16x16x32_bf16(false, a, false, b0, (short)0, acc0, false, false);
            acc1 = __builtin_amdgcn_wmma_f32_16x16x32_bf16(false, a, false, b1, (short)0, acc1, false, false);
            acc2 = __builtin_amdgcn_wmma_f32_16x16x32_bf16(false, a, false, b2, (short)0, acc2, false, false);
            acc3 = __builtin_amdgcn_wmma_f32_16x16x32_bf16(false, a, false, b3, (short)0, acc3, false, false);
        }

        if (chunk + 1 < NCHUNK) {
            asm volatile("s_wait_asynccnt 0x0" ::: "memory");
            __syncthreads();                 // WMMAs done with Bt; raw landed
            transpose_gw(raw, Bt, tid);
            __syncthreads();                 // Bt ready for next chunk
        }
    }

    // ---- spill logits (+bias) to LDS. C/D layout: VGPR i -> M = i + half*8, N = l16
    const float bias0 = gb[ 0 + l16];
    const float bias1 = gb[16 + l16];
    const float bias2 = gb[32 + l16];
    const float bias3 = gb[48 + l16];
    float* lrow = &logitsLds[wave * 16 * LSTR];
    #pragma unroll
    for (int i = 0; i < 8; ++i) {
        const int m = half * 8 + i;
        lrow[m * LSTR +  0 + l16] = acc0[i] + bias0;
        lrow[m * LSTR + 16 + l16] = acc1[i] + bias1;
        lrow[m * LSTR + 32 + l16] = acc2[i] + bias2;
        lrow[m * LSTR + 48 + l16] = acc3[i] + bias3;
    }
    __syncthreads();

    // ---- per-row top-2, renormalized softmax, scatter, aux partials
    if (tid < ROWS_PER_BLOCK) {
        const float* lr = &logitsLds[tid * LSTR];
        const int grow = rowBase + tid;

        float l0 = -INFINITY, l1 = -INFINITY;
        int   i0 = 0, i1 = 0;
        #pragma unroll 8
        for (int e = 0; e < EXPERTS; ++e) {
            const float v = lr[e];
            if (v > l0)      { l1 = l0; i1 = i0; l0 = v; i0 = e; }
            else if (v > l1) { l1 = v;  i1 = e; }
        }
        // softmax over the two selected logits
        const float e1 = __expf(l1 - l0);
        const float s  = 1.0f + e1;
        const float w0 = 1.0f / s;
        const float w1 = e1 / s;

        // full 64-wide softmax for mean_prob_per_expert (row max == l0)
        float denom = 0.0f;
        #pragma unroll 8
        for (int e = 0; e < EXPERTS; ++e) denom += __expf(lr[e] - l0);
        const float inv = 1.0f / denom;
        #pragma unroll 8
        for (int e = 0; e < EXPERTS; ++e)
            atomicAdd(&probSum[e], __expf(lr[e] - l0) * inv);
        atomicAdd(&cntSum[i0], 1.0f);
        atomicAdd(&cntSum[i1], 1.0f);

        // scattered gate weights + selected experts
        float* gwr = gate_weights + (size_t)grow * EXPERTS;
        #pragma unroll 8
        for (int e = 0; e < EXPERTS; ++e)
            gwr[e] = (e == i0) ? w0 : ((e == i1) ? w1 : 0.0f);
        sel_experts[grow * 2 + 0] = i0;
        sel_experts[grow * 2 + 1] = i1;
    }
    __syncthreads();

    // ---- one global atomic per expert per block
    if (tid < EXPERTS) {
        atomicAdd(&ws_prob[tid], probSum[tid]);
        atomicAdd(&ws_cnt[tid],  cntSum[tid]);
    }
}

__global__ void zero_ws_kernel(float* __restrict__ ws)
{
    if (threadIdx.x < 2 * EXPERTS) ws[threadIdx.x] = 0.0f;
}

__global__ void finalize_kernel(const float* __restrict__ ws_prob,
                                const float* __restrict__ ws_cnt,
                                float* __restrict__ aux_out)
{
    __shared__ float red[EXPERTS];
    const int e = threadIdx.x;
    const float frac = ws_cnt[e]  * (1.0f / (float)(TOKENS * 2));
    const float mean = ws_prob[e] * (1.0f / (float)TOKENS);
    red[e] = frac * mean;
    __syncthreads();
    if (e == 0) {
        float s = 0.0f;
        #pragma unroll
        for (int i = 0; i < EXPERTS; ++i) s += red[i];
        *aux_out = (float)EXPERTS * s;
    }
}

extern "C" void kernel_launch(void* const* d_in, const int* in_sizes, int n_in,
                              void* d_out, int out_size, void* d_ws, size_t ws_size,
                              hipStream_t stream)
{
    const float* x  = (const float*)d_in[0];   // [16384, 2048]
    const float* gw = (const float*)d_in[1];   // [2048, 64]
    const float* gb = (const float*)d_in[2];   // [64]

    float* outf         = (float*)d_out;
    float* gate_weights = outf;                                     // [16384, 64] f32
    int*   sel          = (int*)(outf + (size_t)TOKENS * EXPERTS);  // [16384, 2] i32
    float* aux          = outf + (size_t)TOKENS * EXPERTS + (size_t)TOKENS * 2;

    float* wsf = (float*)d_ws;   // [0:64) prob sums, [64:128) counts

    zero_ws_kernel<<<1, 128, 0, stream>>>(wsf);
    moe_gate_kernel<<<TOKENS / ROWS_PER_BLOCK, 256, 0, stream>>>(
        x, gw, gb, gate_weights, sel, wsf, wsf + EXPERTS);
    finalize_kernel<<<1, EXPERTS, 0, stream>>>(wsf, wsf + EXPERTS, aux);
}
